// ESRNN_42399917146763
// MI455X (gfx1250) — matrix operand order repaired and checked by reading, not compiled, not run on hardware
//
#include <hip/hip_runtime.h>
#include <hip/hip_bf16.h>

typedef __attribute__((ext_vector_type(16))) _Float16 v16h;
typedef __attribute__((ext_vector_type(8)))  _Float16 v8h;
typedef __attribute__((ext_vector_type(8)))  float    v8f;

#define B_    256
#define T_    240
#define SEAS_ 12
#define INSZ  12
#define OUTSZ 18
#define HID_  256
#define CATS_ 6
#define NW    229   // T - IN_SZ + 1
#define NWO   211   // T - IN_SZ - OUT_SZ + 1
#define SFULL 264   // T + 2*SEAS

// output layout (flat f32)
#define OFF_NET   0
#define OFF_WOUT  (NWO * B_ * OUTSZ)                 // 972288
#define OFF_HO    (OFF_WOUT + NWO * B_ * OUTSZ)      // 1944576
#define OFF_VAL   (OFF_HO + B_ * OUTSZ)              // 1949184
#define OFF_PEN   (OFF_VAL + B_ * OUTSZ)             // 1953792

// ---- fast transcendentals: gfx1250 v_tanh_f32 (TRANS op).
// ISA §7.3/§8.4: 1 independent op required after a TRANS op before its
// result is consumed -> embed a v_nop in the asm block.
__device__ __forceinline__ float fast_tanh(float x)
{
    float r;
    asm("v_tanh_f32 %0, %1\n\tv_nop" : "=v"(r) : "v"(x));
    return r;
}
__device__ __forceinline__ float fast_sig(float x)
{
    return 0.5f * fast_tanh(0.5f * x) + 0.5f;
}
__device__ __forceinline__ float sigf(float x) { return 1.0f / (1.0f + __expf(-x)); }

// -------- exponential smoothing scan: one thread per series, 1 block --------
__global__ __launch_bounds__(256) void es_kernel(
    const float* __restrict__ train, const float* __restrict__ lev_raw,
    const float* __restrict__ seas_raw, const float* __restrict__ init_seas,
    float* __restrict__ levs, float* __restrict__ seas_full, float* __restrict__ pen_out)
{
    __shared__ float red[256];
    int i = threadIdx.x;
    float ls = sigf(lev_raw[i]);
    float ss = sigf(seas_raw[i]);
    float buf[SEAS_];
#pragma unroll
    for (int s = 0; s < SEAS_; ++s) {
        buf[s] = __expf(init_seas[i * SEAS_ + s]);
        seas_full[i * SFULL + s] = buf[s];
    }
    seas_full[i * SFULL + SEAS_] = buf[0];
    float lev = train[i * T_] / buf[0];
    levs[i * T_] = lev;
    float prev_ld = 0.f, sum = 0.f;
    for (int t = 1; t < T_; ++t) {
        int slot = t % SEAS_;
        float x  = train[i * T_ + t];
        float si = buf[slot];
        float nl = ls * (x / si) + (1.f - ls) * lev;
        float ld = __logf(nl / lev);
        float ns = ss * (x / nl) + (1.f - ss) * si;
        buf[slot] = ns;
        levs[i * T_ + t] = nl;
        seas_full[i * SFULL + t + SEAS_] = ns;   // seas_st col t+12
        if (t >= 2) { float d = ld - prev_ld; sum += d * d; }
        prev_ld = ld;
        lev = nl;
    }
#pragma unroll
    for (int s = 0; s < SEAS_; ++s)
        seas_full[i * SFULL + T_ + SEAS_ + s] = seas_full[i * SFULL + T_ + s];
    red[i] = sum;
    __syncthreads();
    for (int st = 128; st > 0; st >>= 1) {
        if (i < st) red[i] += red[i + st];
        __syncthreads();
    }
    if (i == 0) pen_out[0] = red[0] / (float)((T_ - 2) * B_);
}

// -------- window construction --------
__global__ void win_in_kernel(const float* __restrict__ train, const float* __restrict__ seas,
                              const float* __restrict__ levs, const float* __restrict__ info_cat,
                              _Float16* __restrict__ x0)
{
    int w = blockIdx.x, i = threadIdx.x;
    float lv = levs[i * T_ + INSZ - 1 + w];
    size_t base = ((size_t)w * B_ + i) * 32;
#pragma unroll
    for (int k = 0; k < INSZ; ++k)
        x0[base + k] = (_Float16)(train[i * T_ + w + k] / seas[i * SFULL + w + k] / lv);
#pragma unroll
    for (int k = 0; k < CATS_; ++k)
        x0[base + INSZ + k] = (_Float16)info_cat[i * CATS_ + k];
#pragma unroll
    for (int k = INSZ + CATS_; k < 32; ++k) x0[base + k] = (_Float16)0.f;
}

__global__ void win_out_kernel(const float* __restrict__ train, const float* __restrict__ seas,
                               const float* __restrict__ levs, float* __restrict__ wout)
{
    int w = blockIdx.x, i = threadIdx.x;   // w < NWO
    float lv = levs[i * T_ + INSZ - 1 + w];
#pragma unroll
    for (int k = 0; k < OUTSZ; ++k) {
        int tt = INSZ + w + k;
        wout[((size_t)w * B_ + i) * OUTSZ + k] = train[i * T_ + tt] / seas[i * SFULL + tt] / lv;
    }
}

// -------- weight packers: B-operand fragment layout --------
// fragment flat index = frag*512 + lane*16 + half, frag = kt*NT + nt
// n = nt*16 + lane%16 ; k = kt*32 + (lane<16 ? 0 : 8) + (half&7) + (half<8 ? 0 : 16)
__global__ void pack_lstm_w(const float* __restrict__ Wih, const float* __restrict__ Whh,
                            _Float16* __restrict__ Wc, int d, int dp)
{
    int idx = blockIdx.x * blockDim.x + threadIdx.x;
    int Kp = dp + HID_;
    if (idx >= Kp * 1024) return;
    int half = idx & 15, lane = (idx >> 4) & 31, frag = idx >> 9;
    int kt = frag >> 6, nt = frag & 63;
    int n = nt * 16 + (lane & 15);
    int k = kt * 32 + ((lane < 16) ? 0 : 8) + (half & 7) + ((half < 8) ? 0 : 16);
    float v = 0.f;
    if (k < dp) { if (k < d) v = Wih[n * d + k]; }
    else        { v = Whh[n * HID_ + (k - dp)]; }
    Wc[idx] = (_Float16)v;
}

__global__ void pack_w(const float* __restrict__ W, _Float16* __restrict__ out,
                       int N, int K, int Npad)
{
    int NT = Npad / 16;
    int total = (K / 32) * NT * 512;
    int idx = blockIdx.x * blockDim.x + threadIdx.x;
    if (idx >= total) return;
    int half = idx & 15, lane = (idx >> 4) & 31, frag = idx >> 9;
    int kt = frag / NT, nt = frag % NT;
    int n = nt * 16 + (lane & 15);
    int k = kt * 32 + ((lane < 16) ? 0 : 8) + (half & 7) + ((half < 8) ? 0 : 16);
    float v = (n < N) ? W[n * K + k] : 0.f;
    out[idx] = (_Float16)v;
}

__global__ void pack_bias(const float* __restrict__ bih, const float* __restrict__ bhh,
                          float* __restrict__ bc)
{
    int n = blockIdx.x * blockDim.x + threadIdx.x;
    if (n < 1024) bc[n] = bih[n] + bhh[n];
}

__global__ void pad_scb(const float* __restrict__ scb, float* __restrict__ scbp)
{
    int n = threadIdx.x;
    if (n < 32) scbp[n] = (n < OUTSZ) ? scb[n] : 0.f;
}

// -------- fragment loaders --------
__device__ __forceinline__ v16h ldfrag(const _Float16* p)
{
    v8h lo = *(const v8h*)p;
    v8h hi = *(const v8h*)(p + 16);
    v16h r;
#pragma unroll
    for (int q = 0; q < 8; ++q) { r[q] = lo[q]; r[q + 8] = hi[q]; }
    return r;
}

__device__ __forceinline__ void mm8(v16h a, const _Float16* __restrict__ Wc,
                                    int kt, int wid, int lane, v8f acc[2][4])
{
#pragma unroll
    for (int g = 0; g < 2; ++g)
#pragma unroll
        for (int q = 0; q < 4; ++q) {
            int tile = wid + 8 * g + 16 * q;            // i/f/g/o bands per hidden col group
            v16h b = *(const v16h*)(Wc + ((size_t)(kt * 64 + tile)) * 512 + (size_t)lane * 16);
            acc[g][q] = __builtin_amdgcn_wmma_f32_16x16x32_f16(
                false, a, false, b, (short)0, acc[g][q], false, false);
        }
}

// -------- parallel input-gate GEMM: Xg = bias + X @ [Wih]^T for all rows --------
// Output fragments in C-layout order: frag = (rowtile*64 + tile), element
// offset lane*8 + v (256 f32 per 16x16 tile). Xg is a 240MB single-use
// stream -> non-temporal stores keep weights/H resident in L2.
template <int KXT>
__global__ __launch_bounds__(256) void xg_gemm(
    const _Float16* __restrict__ X,   // (T, B, dp) f16
    const _Float16* __restrict__ Wc,  // packed fragments (X part = kt 0..KXT-1)
    const float*    __restrict__ bc,
    float*          __restrict__ Xg)
{
    const int dp = KXT * 32;
    int tid = threadIdx.x, lane = tid & 31, wid = tid >> 5;
    int m = lane & 15, aoff = (lane < 16) ? 0 : 8;
    int rt = blockIdx.x;                 // row-tile in natural (t,i) order
    int t  = rt >> 4;
    int i0 = (rt & 15) << 4;

    v8f acc[2][4];
#pragma unroll
    for (int g = 0; g < 2; ++g)
#pragma unroll
        for (int q = 0; q < 4; ++q) {
            float bv = bc[(wid + 8 * g + 16 * q) * 16 + m];
#pragma unroll
            for (int v = 0; v < 8; ++v) acc[g][q][v] = bv;
        }
    const _Float16* xrow = X + ((size_t)t * B_ + (i0 + m)) * dp;
#pragma unroll
    for (int kt = 0; kt < KXT; ++kt) {
        v16h a = ldfrag(xrow + kt * 32 + aoff);
        mm8(a, Wc, kt, wid, lane, acc);
    }
#pragma unroll
    for (int g = 0; g < 2; ++g)
#pragma unroll
        for (int q = 0; q < 4; ++q) {
            int tile = wid + 8 * g + 16 * q;
            v8f* p = (v8f*)(Xg + ((size_t)rt * 64 + tile) * 256 + (size_t)lane * 8);
            __builtin_nontemporal_store(acc[g][q], p);
        }
}

// -------- dilated LSTM layer: one WG owns 16 batch rows, all 1024 gates --------
// XG=true : input-gate contribution precomputed (acc init = Xg fragment),
//           sequential K-loop covers only the 8 hidden K-tiles.
// XG=false: fused fallback (X read from global inside the step loop).
template <int KXT, bool XG>
__global__ __launch_bounds__(256) void lstm_layer(
    const _Float16* __restrict__ X,    // (T, B, dp) f16 (fused mode)
    const float*    __restrict__ Xg,   // precomputed gate fragments (XG mode)
    const _Float16* __restrict__ Wc,   // packed fragments, Kp = dp + 256
    const float*    __restrict__ bc,   // bih + bhh (1024)
    _Float16*       __restrict__ Hout, // (T, B, 256) f16
    int rate, int steps, int Treal)
{
    const int dp = KXT * 32;
    __shared__ _Float16 hbuf[2][16 * HID_];
    int tid = threadIdx.x, lane = tid & 31, wid = tid >> 5;
    for (int q = tid; q < 16 * HID_; q += 256) hbuf[0][q] = (_Float16)0.f;
    __syncthreads();

    int gr0 = blockIdx.x * 16;        // 16 rows of the (rate*B) scan batch
    int j   = gr0 >> 8;               // dilation phase (16 | 256 so uniform in WG)
    int i0  = gr0 & (B_ - 1);
    int m    = lane & 15;
    int aoff = (lane < 16) ? 0 : 8;

    // bias hoisted to registers (used every step / for padded steps)
    float bias_r[2][4];
#pragma unroll
    for (int g = 0; g < 2; ++g)
#pragma unroll
        for (int q = 0; q < 4; ++q)
            bias_r[g][q] = bc[(wid + 8 * g + 16 * q) * 16 + m];

    v8f c[2], acc[2][4];
#pragma unroll
    for (int g = 0; g < 2; ++g)
#pragma unroll
        for (int v = 0; v < 8; ++v) c[g][v] = 0.f;

    for (int k = 0; k < steps; ++k) {
        int t = k * rate + j;
        bool valid = t < Treal;       // uniform across WG
        int cur = k & 1, nxt = cur ^ 1;

        if (XG && valid) {
            size_t rt = (size_t)t * 16 + (i0 >> 4);
#pragma unroll
            for (int g = 0; g < 2; ++g)
#pragma unroll
                for (int q = 0; q < 4; ++q) {
                    int tile = wid + 8 * g + 16 * q;
                    const v8f* p = (const v8f*)(Xg + (rt * 64 + tile) * 256 + (size_t)lane * 8);
                    acc[g][q] = __builtin_nontemporal_load(p);
                }
            // prefetch next step's gate fragments (global_prefetch_b8: no
            // counter, no VGPR return) so they ride under this step's WMMAs
            int tn = t + rate;
            if (tn < Treal) {
                size_t rtn = (size_t)tn * 16 + (i0 >> 4);
#pragma unroll
                for (int g = 0; g < 2; ++g)
#pragma unroll
                    for (int q = 0; q < 4; ++q) {
                        int tile = wid + 8 * g + 16 * q;
                        __builtin_prefetch(Xg + (rtn * 64 + tile) * 256 + (size_t)lane * 8, 0, 1);
                    }
            }
        } else {
#pragma unroll
            for (int g = 0; g < 2; ++g)
#pragma unroll
                for (int q = 0; q < 4; ++q) {
                    float bv = bias_r[g][q];
#pragma unroll
                    for (int v = 0; v < 8; ++v) acc[g][q][v] = bv;
                }
            if (!XG && valid) {
                const _Float16* xrow = X + ((size_t)t * B_ + (i0 + m)) * dp;
#pragma unroll
                for (int kt = 0; kt < KXT; ++kt) {
                    v16h a = ldfrag(xrow + kt * 32 + aoff);
                    mm8(a, Wc, kt, wid, lane, acc);
                }
            }
        }

        // recurrent contribution from LDS h
#pragma unroll
        for (int kh = 0; kh < HID_ / 32; ++kh) {
            v16h a = ldfrag(&hbuf[cur][m * HID_ + kh * 32 + aoff]);
            mm8(a, Wc, KXT + kh, wid, lane, acc);
        }

        // LSTM activations: i,f,g,o live in the same wave for each hidden col
#pragma unroll
        for (int g = 0; g < 2; ++g) {
            int col = (wid + 8 * g) * 16 + m;
#pragma unroll
            for (int v = 0; v < 8; ++v) {
                float ig = fast_sig(acc[g][0][v]);
                float fg = fast_sig(acc[g][1][v]);
                float gg = fast_tanh(acc[g][2][v]);
                float og = fast_sig(acc[g][3][v]);
                float cc = fg * c[g][v] + ig * gg;
                c[g][v] = cc;
                float h = og * fast_tanh(cc);
                int row = v + ((lane < 16) ? 0 : 8);
                hbuf[nxt][row * HID_ + col] = (_Float16)h;
                if (valid)
                    Hout[((size_t)t * B_ + (i0 + row)) * HID_ + col] = (_Float16)h;
            }
        }
        __syncthreads();
    }
}

// -------- head: out = tanh((H1+H3) @ nlW^T + nlb) @ scW^T + scb --------
__global__ __launch_bounds__(256) void head_kernel(
    const _Float16* __restrict__ O1, const _Float16* __restrict__ O3,
    const _Float16* __restrict__ nlWp, const float* __restrict__ nlb,
    const _Float16* __restrict__ scWp, const float* __restrict__ scbp,
    const float* __restrict__ seas, const float* __restrict__ levs,
    float* __restrict__ net, float* __restrict__ ho)
{
    __shared__ _Float16 sbuf[16 * HID_];
    __shared__ _Float16 zbuf[16 * HID_];
    int tid = threadIdx.x, lane = tid & 31, wid = tid >> 5, m = lane & 15;
    int gr0 = blockIdx.x * 16;
    int t = gr0 >> 8, i0 = gr0 & (B_ - 1);
    {
        int r  = tid >> 4;
        int c0 = (tid & 15) * 16;
        size_t base = ((size_t)gr0 + r) * HID_ + c0;
#pragma unroll
        for (int q = 0; q < 16; ++q)
            sbuf[r * HID_ + c0 + q] = (_Float16)((float)O1[base + q] + (float)O3[base + q]);
    }
    __syncthreads();
    int aoff = (lane < 16) ? 0 : 8;

    v8f zacc[2];
#pragma unroll
    for (int g = 0; g < 2; ++g) {
        float bv = nlb[(wid + 8 * g) * 16 + m];
#pragma unroll
        for (int v = 0; v < 8; ++v) zacc[g][v] = bv;
    }
#pragma unroll
    for (int kt = 0; kt < HID_ / 32; ++kt) {
        v16h a = ldfrag(&sbuf[m * HID_ + kt * 32 + aoff]);
#pragma unroll
        for (int g = 0; g < 2; ++g) {
            int nt = wid + 8 * g;
            v16h b = *(const v16h*)(nlWp + ((size_t)(kt * 16 + nt)) * 512 + (size_t)lane * 16);
            zacc[g] = __builtin_amdgcn_wmma_f32_16x16x32_f16(
                false, a, false, b, (short)0, zacc[g], false, false);
        }
    }
#pragma unroll
    for (int g = 0; g < 2; ++g) {
        int col = (wid + 8 * g) * 16 + m;
#pragma unroll
        for (int v = 0; v < 8; ++v) {
            int row = v + ((lane < 16) ? 0 : 8);
            zbuf[row * HID_ + col] = (_Float16)fast_tanh(zacc[g][v]);
        }
    }
    __syncthreads();

    if (wid < 2) {   // wave-uniform branch: EXEC stays all-ones inside
        int nt = wid;
        v8f pacc;
        float bv = scbp[nt * 16 + m];
#pragma unroll
        for (int v = 0; v < 8; ++v) pacc[v] = bv;
#pragma unroll
        for (int kt = 0; kt < HID_ / 32; ++kt) {
            v16h a = ldfrag(&zbuf[m * HID_ + kt * 32 + aoff]);
            v16h b = *(const v16h*)(scWp + ((size_t)(kt * 2 + nt)) * 512 + (size_t)lane * 16);
            pacc = __builtin_amdgcn_wmma_f32_16x16x32_f16(
                false, a, false, b, (short)0, pacc, false, false);
        }
        int col = nt * 16 + m;
        if (col < OUTSZ) {
#pragma unroll
            for (int v = 0; v < 8; ++v) {
                int row = v + ((lane < 16) ? 0 : 8);
                int i = i0 + row;
                float val = pacc[v];
                if (t < NWO)
                    net[((size_t)t * B_ + i) * OUTSZ + col] = val;
                if (t == NW - 1) {
                    float hv = val * seas[i * SFULL + (SFULL - OUTSZ) + col] * levs[i * T_ + T_ - 1];
                    ho[i * OUTSZ + col] = hv > 0.f ? hv : 0.f;
                }
            }
        }
    }
}

extern "C" void kernel_launch(void* const* d_in, const int* in_sizes, int n_in,
                              void* d_out, int out_size, void* d_ws, size_t ws_size,
                              hipStream_t stream)
{
    const float* Wih[4] = {(const float*)d_in[0], (const float*)d_in[4], (const float*)d_in[8],  (const float*)d_in[12]};
    const float* Whh[4] = {(const float*)d_in[1], (const float*)d_in[5], (const float*)d_in[9],  (const float*)d_in[13]};
    const float* bih[4] = {(const float*)d_in[2], (const float*)d_in[6], (const float*)d_in[10], (const float*)d_in[14]};
    const float* bhh[4] = {(const float*)d_in[3], (const float*)d_in[7], (const float*)d_in[11], (const float*)d_in[15]};
    const float* nlW = (const float*)d_in[16];
    const float* nlb = (const float*)d_in[17];
    const float* scW = (const float*)d_in[18];
    const float* scb = (const float*)d_in[19];
    const float* train = (const float*)d_in[20];
    const float* val   = (const float*)d_in[21];
    const float* info_cat = (const float*)d_in[23];
    const float* lev_raw  = (const float*)d_in[24];
    const float* seas_raw = (const float*)d_in[25];
    const float* init_seas = (const float*)d_in[26];
    float* out = (float*)d_out;

    // workspace carve-up (256B aligned)
    size_t off = 0;
    auto carve = [&](size_t bytes) -> void* {
        void* p = (char*)d_ws + off;
        off += (bytes + 255) & ~(size_t)255;
        return p;
    };
    float*     seas  = (float*)carve((size_t)B_ * SFULL * 4);
    float*     levs  = (float*)carve((size_t)B_ * T_ * 4);
    float*     scbp  = (float*)carve(32 * 4);
    float*     bc    = (float*)carve(4 * 1024 * 4);
    _Float16*  x0    = (_Float16*)carve((size_t)NW * B_ * 32 * 2);
    _Float16*  Wc[4];
    Wc[0] = (_Float16*)carve((size_t)(32 + HID_) * 1024 * 2);
    for (int l = 1; l < 4; ++l) Wc[l] = (_Float16*)carve((size_t)(HID_ + HID_) * 1024 * 2);
    _Float16*  nlWp  = (_Float16*)carve((size_t)HID_ * HID_ * 2);
    _Float16*  scWp  = (_Float16*)carve((size_t)8 * 2 * 512 * 2);
    _Float16*  H[4];
    for (int l = 0; l < 4; ++l) H[l] = (_Float16*)carve((size_t)NW * B_ * HID_ * 2);
    // precomputed gate fragments (f32), reused across layers: 240 MB
    float* Xg = (float*)carve((size_t)NW * B_ * 1024 * 4);
    bool use_xg = (off <= ws_size);

    // ES scan + pen
    es_kernel<<<1, 256, 0, stream>>>(train, lev_raw, seas_raw, init_seas,
                                     levs, seas, out + OFF_PEN);
    // windows
    win_in_kernel<<<NW, 256, 0, stream>>>(train, seas, levs, info_cat, x0);
    win_out_kernel<<<NWO, 256, 0, stream>>>(train, seas, levs, out + OFF_WOUT);

    // weight packing
    const int d_l[4]  = {INSZ + CATS_, HID_, HID_, HID_};
    const int dp_l[4] = {32, HID_, HID_, HID_};
    for (int l = 0; l < 4; ++l) {
        int halves = (dp_l[l] + HID_) * 1024;
        pack_lstm_w<<<(halves + 255) / 256, 256, 0, stream>>>(Wih[l], Whh[l], Wc[l], d_l[l], dp_l[l]);
        pack_bias<<<4, 256, 0, stream>>>(bih[l], bhh[l], bc + l * 1024);
    }
    pack_w<<<(8 * 16 * 512) / 256, 256, 0, stream>>>(nlW, nlWp, HID_, HID_, HID_);
    pack_w<<<(8 * 2 * 512) / 256, 256, 0, stream>>>(scW, scWp, OUTSZ, HID_, 32);
    pad_scb<<<1, 32, 0, stream>>>(scb, scbp);

    const int RT  = NW * B_ / 16;                     // 3664 row-tiles
    const int stp[4] = {NW, (NW + 2) / 3, (NW + 5) / 6, (NW + 11) / 12};
    const int rat[4] = {1, 3, 6, 12};

    if (use_xg) {
        // layer 0
        xg_gemm<1><<<RT, 256, 0, stream>>>(x0, Wc[0], bc, Xg);
        lstm_layer<1, true><<<rat[0] * B_ / 16, 256, 0, stream>>>(
            nullptr, Xg, Wc[0], bc, H[0], rat[0], stp[0], NW);
        // layers 1..3: parallel input GEMM feeds the halved sequential loop
        for (int l = 1; l < 4; ++l) {
            xg_gemm<8><<<RT, 256, 0, stream>>>(H[l - 1], Wc[l], bc + l * 1024, Xg);
            lstm_layer<8, true><<<rat[l] * B_ / 16, 256, 0, stream>>>(
                nullptr, Xg, Wc[l], bc + l * 1024, H[l], rat[l], stp[l], NW);
        }
    } else {
        lstm_layer<1, false><<<rat[0] * B_ / 16, 256, 0, stream>>>(
            x0, nullptr, Wc[0], bc, H[0], rat[0], stp[0], NW);
        for (int l = 1; l < 4; ++l)
            lstm_layer<8, false><<<rat[l] * B_ / 16, 256, 0, stream>>>(
                H[l - 1], nullptr, Wc[l], bc + l * 1024, H[l], rat[l], stp[l], NW);
    }

    // head: residual H1 + H3 -> tanh MLP -> scoring; writes net_pred + ho_pred
    head_kernel<<<RT, 256, 0, stream>>>(H[1], H[3], nlWp, nlb, scWp, scbp,
                                        seas, levs, out + OFF_NET, out + OFF_HO);

    // val passthrough
    hipMemcpyAsync(out + OFF_VAL, val, (size_t)B_ * OUTSZ * 4, hipMemcpyDeviceToDevice, stream);
}
